// Encoder_79920751444202
// MI455X (gfx1250) — matrix operand order repaired and checked by reading, not compiled, not run on hardware
//
#include <hip/hip_runtime.h>

typedef __bf16 bf16;
typedef __attribute__((ext_vector_type(16))) __bf16 v16bf;
typedef __attribute__((ext_vector_type(8)))  float  v8f;

static constexpr int T_STEPS = 1024;
static constexpr int BATCH   = 256;
static constexpr int HID     = 64;
static constexpr int GDIM    = 192;   // 3*H
static constexpr int WPAD    = 8;     // halves of pad per weight row (bank step 4)

// 16B-aligned v16bf load (weight rows are 16B- but not 32B-aligned)
__device__ __forceinline__ v16bf load_v16bf(const bf16* p) {
    union { uint4 q[2]; v16bf v; } u;
    u.q[0] = *(const uint4*)p;
    u.q[1] = *(const uint4*)(p + 8);
    return u.v;
}

// ---------------------------------------------------------------------------
// mfcc (T,B,64) f32 -> A-fragment-swizzled bf16 tiles.
// Tile (t, bt) holds 16 rows x 64 cols; half index within tile:
//   kg*512 + (g*16 + m)*16 + e   with  k = kg*32 + 16*g + e
__global__ __launch_bounds__(256) void swizzle_x0_kernel(
    const float* __restrict__ src, bf16* __restrict__ dst, int nruns)
{
    int r = blockIdx.x * blockDim.x + threadIdx.x;
    if (r >= nruns) return;
    int eh   = r & 1;
    int m    = (r >> 1) & 15;
    int g    = (r >> 5) & 1;
    int kg   = (r >> 6) & 1;
    int tile = r >> 7;           // t*16 + bt
    int t    = tile >> 4;
    int b    = (tile & 15) * 16 + m;
    int k0   = kg * 32 + 16 * g + 8 * eh;
    const float* s = src + ((size_t)t * BATCH + b) * 64 + k0;
    __align__(16) bf16 tmp[8];
    #pragma unroll
    for (int i = 0; i < 8; ++i) tmp[i] = (bf16)s[i];
    *(uint4*)(dst + (size_t)r * 8) = *(const uint4*)tmp;
}

// ---------------------------------------------------------------------------
template<int KIN>
struct Smem {
    static constexpr int WIH_STRIDE = KIN + WPAD;   // halves
    static constexpr int WHH_STRIDE = HID + WPAD;
    static constexpr int o_wih = 0;
    static constexpr int o_whh = o_wih + GDIM * WIH_STRIDE * 2;
    static constexpr int o_hA  = o_whh + GDIM * WHH_STRIDE * 2;     // 2 buffers
    static constexpr int total = o_hA + 2 * 16 * HID * 2;
};

// One (batch-tile, direction) per block; 128 threads = 4 waves; scans all T.
// Wave w owns output columns {w*16, 64+w*16, 128+w*16}; r/z use a fused
// gi+gh accumulator (WMMA C-chaining), n keeps gi/gh separate.
template<int KIN>
__global__ __launch_bounds__(128) void gru_layer_kernel(
    const bf16*  __restrict__ x,      // swizzled (T*16) tiles of 16*KIN halves
    const float* __restrict__ WihG,   // (2, GDIM, KIN)
    const float* __restrict__ WhhG,   // (2, GDIM, HID)
    const float* __restrict__ bihG,   // (2, GDIM)
    const float* __restrict__ bhhG,   // (2, GDIM)
    bf16*        __restrict__ y,      // swizzled (T*16) tiles of 16*128, or null
    float*       __restrict__ hT_d0,  // final h, stride 128, or null
    float*       __restrict__ hT_d1)
{
    using L = Smem<KIN>;
    constexpr int KFRAG = KIN / 32;
    extern __shared__ char smem[];
    bf16* s_wih = (bf16*)(smem + L::o_wih);
    bf16* s_whh = (bf16*)(smem + L::o_whh);
    bf16* s_hA  = (bf16*)(smem + L::o_hA);

    const int tid  = threadIdx.x;
    const int lane = tid & 31;
    const int wave = tid >> 5;
    const int bt   = blockIdx.x & 15;
    const int dir  = blockIdx.x >> 4;
    const bool rev = (dir == 1);
    const int m    = lane & 15;
    const int gsel = lane >> 4;
    const int rbase = (lane < 16) ? 0 : 8;

    const float* Wih = WihG + (size_t)dir * GDIM * KIN;
    const float* Whh = WhhG + (size_t)dir * GDIM * HID;

    // weights -> padded bf16 LDS (once); zero both h ping-pong buffers
    for (int i = tid; i < GDIM * KIN; i += 128)
        s_wih[(i / KIN) * L::WIH_STRIDE + (i % KIN)] = (bf16)Wih[i];
    for (int i = tid; i < GDIM * HID; i += 128)
        s_whh[(i / HID) * L::WHH_STRIDE + (i % HID)] = (bf16)Whh[i];
    for (int i = tid; i < 2 * 16 * HID; i += 128) s_hA[i] = (bf16)0.0f;

    // scalar gate biases for this lane's column (folded into gate math)
    const int jcol = wave * 16 + m;
    const float brz0 = bihG[dir * GDIM + jcol]       + bhhG[dir * GDIM + jcol];
    const float brz1 = bihG[dir * GDIM + 64 + jcol]  + bhhG[dir * GDIM + 64 + jcol];
    const float bi2  = bihG[dir * GDIM + 128 + jcol];
    const float bh2  = bhhG[dir * GDIM + 128 + jcol];

    // h slice in registers: h[rbase+v][jcol]
    v8f hreg;
    #pragma unroll
    for (int v = 0; v < 8; ++v) hreg[v] = 0.0f;

    // A-layout position of this lane's h column (for the bf16 h stores)
    const int hkg  = jcol >> 5;
    const int hkr  = jcol & 31;
    const int hg   = (hkr >> 4) & 1;
    const int he   = hkr & 15;
    const int hbase = hkg * 512 + (hg * 16 + rbase) * 16 + he;

    __syncthreads();

    const v8f vzero = (v8f)0.0f;   // folds to inline C=0 in WMMA

    // prologue: x fragments for the first step
    v16bf aX0[KFRAG], aX1[KFRAG];
    {
        const int tfirst = rev ? (T_STEPS - 1) : 0;
        const bf16* xt = x + (size_t)(tfirst * 16 + bt) * (16 * KIN);
        #pragma unroll
        for (int kg = 0; kg < KFRAG; ++kg)
            aX0[kg] = load_v16bf(xt + kg * 512 + lane * 16);
    }

    auto step = [&](int s, v16bf (&aX)[KFRAG], v16bf (&aXn)[KFRAG]) {
        const int t = rev ? (T_STEPS - 1 - s) : s;
        int tn = rev ? (T_STEPS - 2 - s) : (s + 1);
        if (tn < 0) tn = 0;
        if (tn >= T_STEPS) tn = T_STEPS - 1;

        // prefetch next step's x fragments (full iteration of latency)
        {
            const bf16* xt = x + (size_t)(tn * 16 + bt) * (16 * KIN);
            #pragma unroll
            for (int kg = 0; kg < KFRAG; ++kg)
                aXn[kg] = load_v16bf(xt + kg * 512 + lane * 16);
        }

        // h fragments from current ping-pong buffer
        const bf16* hbuf = s_hA + (s & 1) * (16 * HID);
        v16bf aH[2];
        #pragma unroll
        for (int kg = 0; kg < 2; ++kg)
            aH[kg] = load_v16bf(hbuf + kg * 512 + lane * 16);

        // --- r/z tiles: fused gi+gh accumulation from inline-zero C ---
        v8f accRZ[2];
        #pragma unroll
        for (int i = 0; i < 2; ++i) {
            const int n = i * 64 + jcol;
            v8f acc = vzero;
            #pragma unroll
            for (int kg = 0; kg < KFRAG; ++kg) {
                v16bf b = load_v16bf(s_wih + n * L::WIH_STRIDE + kg * 32 + gsel * 16);
                acc = __builtin_amdgcn_wmma_f32_16x16x32_bf16(
                    false, aX[kg], false, b, (short)0, acc, false, false);
            }
            #pragma unroll
            for (int kg = 0; kg < 2; ++kg) {
                v16bf b = load_v16bf(s_whh + n * L::WHH_STRIDE + kg * 32 + gsel * 16);
                acc = __builtin_amdgcn_wmma_f32_16x16x32_bf16(
                    false, aH[kg], false, b, (short)0, acc, false, false);
            }
            accRZ[i] = acc;
        }
        // --- n tile: gi and gh kept separate (r multiplies only gh) ---
        v8f accI2 = vzero, accH2 = vzero;
        {
            const int n = 128 + jcol;
            #pragma unroll
            for (int kg = 0; kg < KFRAG; ++kg) {
                v16bf b = load_v16bf(s_wih + n * L::WIH_STRIDE + kg * 32 + gsel * 16);
                accI2 = __builtin_amdgcn_wmma_f32_16x16x32_bf16(
                    false, aX[kg], false, b, (short)0, accI2, false, false);
            }
            #pragma unroll
            for (int kg = 0; kg < 2; ++kg) {
                v16bf b = load_v16bf(s_whh + n * L::WHH_STRIDE + kg * 32 + gsel * 16);
                accH2 = __builtin_amdgcn_wmma_f32_16x16x32_bf16(
                    false, aH[kg], false, b, (short)0, accH2, false, false);
            }
        }

        // fused gate math in registers; emit new h (bf16, A-layout)
        bf16* hout = s_hA + ((s + 1) & 1) * (16 * HID);
        #pragma unroll
        for (int v = 0; v < 8; ++v) {
            float r  = 1.0f / (1.0f + __expf(-(accRZ[0][v] + brz0)));
            float z  = 1.0f / (1.0f + __expf(-(accRZ[1][v] + brz1)));
            float pn = (accI2[v] + bi2) + r * (accH2[v] + bh2);
            float nn;
            asm volatile("v_tanh_f32 %0, %1" : "=v"(nn) : "v"(pn));   // HW tanh
            float h = (1.0f - z) * nn + z * hreg[v];
            hreg[v] = h;
            hout[hbase + v * 16] = (bf16)h;
        }

        __syncthreads();   // the only barrier per step

        // stream y tile: new h buffer is already the consumer's A-layout
        if (y) {
            bf16* dstb = y + (size_t)(t * 16 + bt) * (16 * 128) + dir * (16 * HID);
            *(uint4*)(dstb + tid * 8) = *(const uint4*)(hout + tid * 8);
        }
    };

    // 2x unrolled scan with ping-pong x fragment registers (no copies)
    for (int s = 0; s < T_STEPS; s += 2) {
        step(s,     aX0, aX1);
        step(s + 1, aX1, aX0);
    }

    float* hT = (dir == 0) ? hT_d0 : hT_d1;
    if (hT) {
        #pragma unroll
        for (int v = 0; v < 8; ++v)
            hT[(size_t)(bt * 16 + rbase + v) * 128 + jcol] = hreg[v];
    }
}

// ---------------------------------------------------------------------------
extern "C" void kernel_launch(void* const* d_in, const int* in_sizes, int n_in,
                              void* d_out, int out_size, void* d_ws, size_t ws_size,
                              hipStream_t stream)
{
    (void)in_sizes; (void)n_in; (void)out_size; (void)ws_size;

    const float* mfcc = (const float*)d_in[0];   // (1024, 256, 64)
    const float* Wih0 = (const float*)d_in[1];   // (2, 192, 64)
    const float* Wih  = (const float*)d_in[2];   // (3, 2, 192, 128)
    const float* Whh  = (const float*)d_in[3];   // (4, 2, 192, 64)
    const float* bih  = (const float*)d_in[4];   // (4, 2, 192)
    const float* bhh  = (const float*)d_in[5];   // (4, 2, 192)
    float*       out  = (float*)d_out;           // (256, 128)

    char* ws = (char*)d_ws;
    bf16* Xswz = (bf16*)ws;                                           // T*B*64
    size_t off = (size_t)T_STEPS * BATCH * 64 * sizeof(bf16);
    bf16* buf1 = (bf16*)(ws + off);                                   // T*B*128
    off += (size_t)T_STEPS * BATCH * 128 * sizeof(bf16);
    bf16* buf2 = (bf16*)(ws + off);                                   // T*B*128

    // mfcc -> swizzled bf16 A-fragment tiles
    int nruns = T_STEPS * BATCH * 64 / 8;
    swizzle_x0_kernel<<<(nruns + 255) / 256, 256, 0, stream>>>(mfcc, Xswz, nruns);

    const size_t lds64  = Smem<64>::total;
    const size_t lds128 = Smem<128>::total;

    // Layer 0 (KIN=64), both directions -> buf1
    gru_layer_kernel<64><<<dim3(32), dim3(128), lds64, stream>>>(
        Xswz, Wih0, Whh, bih, bhh, buf1, nullptr, nullptr);

    // Layer 1 (KIN=128), both directions -> buf2; reverse final h -> out[:, 0:64]
    gru_layer_kernel<128><<<dim3(32), dim3(128), lds128, stream>>>(
        buf1, Wih,
        Whh + 2 * GDIM * HID, bih + 2 * GDIM, bhh + 2 * GDIM,
        buf2, nullptr, out);

    // Layer 2 (KIN=128), forward only; final h -> out[:, 64:128]; no y needed
    gru_layer_kernel<128><<<dim3(16), dim3(128), lds128, stream>>>(
        buf2, Wih + 2 * GDIM * 128,
        Whh + 4 * GDIM * HID, bih + 4 * GDIM, bhh + 4 * GDIM,
        nullptr, out + 64, nullptr);
}
